// MambaLMHeadModel_57526791963264
// MI455X (gfx1250) — matrix operand order repaired and checked by reading, not compiled, not run on hardware
//
#include <hip/hip_runtime.h>
#include <hip/hip_bf16.h>

typedef __attribute__((ext_vector_type(16))) _Float16 v16h;
typedef __attribute__((ext_vector_type(8)))  float    v8f;

#define D_MODEL 768
#define D_INNER 1536
#define D_STATE 16
#define DT_RANK 48
#define SEQ_L   1024

#define MT 2   // 16-row tiles per wave
#define NT 4   // 16-col tiles per wave

// ---------------------------------------------------------------------------
// Embedding gather: h[l][:] = embedding[ids[l]][:]
// ---------------------------------------------------------------------------
__global__ __launch_bounds__(256) void k_embed(const int* __restrict__ ids,
                                               const float* __restrict__ emb,
                                               float* __restrict__ h) {
    int l = blockIdx.x;
    const float* src = emb + (size_t)ids[l] * D_MODEL;
    float* dst = h + (size_t)l * D_MODEL;
    for (int d = threadIdx.x; d < D_MODEL; d += 256) dst[d] = src[d];
}

// ---------------------------------------------------------------------------
// RMSNorm, one block per row of length D
// ---------------------------------------------------------------------------
__global__ __launch_bounds__(256) void k_rmsnorm(const float* __restrict__ x,
                                                 const float* __restrict__ w,
                                                 float* __restrict__ o, int D) {
    int l = blockIdx.x;
    const float* xr = x + (size_t)l * D;
    float ss = 0.f;
    for (int d = threadIdx.x; d < D; d += 256) { float v = xr[d]; ss += v * v; }
    __shared__ float red[256];
    red[threadIdx.x] = ss;
    __syncthreads();
    for (int s = 128; s > 0; s >>= 1) {
        if (threadIdx.x < s) red[threadIdx.x] += red[threadIdx.x + s];
        __syncthreads();
    }
    float rs = rsqrtf(red[0] / (float)D + 1e-5f);
    float* orow = o + (size_t)l * D;
    for (int d = threadIdx.x; d < D; d += 256) orow[d] = xr[d] * rs * w[d];
}

// ---------------------------------------------------------------------------
// Causal depthwise conv (k=4) + bias + SiLU.  x = xz[:, 0:D_INNER]
// ---------------------------------------------------------------------------
__global__ __launch_bounds__(256) void k_conv_silu(const float* __restrict__ xz,
                                                   const float* __restrict__ cw,
                                                   const float* __restrict__ cb,
                                                   float* __restrict__ xc) {
    int id = blockIdx.x * 256 + threadIdx.x;
    if (id >= SEQ_L * D_INNER) return;
    int c = id % D_INNER, l = id / D_INNER;
    float s = cb[c];
#pragma unroll
    for (int j = 0; j < 4; ++j) {
        int ll = l - 3 + j;
        if (ll >= 0) s += cw[c * 4 + j] * xz[(size_t)ll * (2 * D_INNER) + c];
    }
    float sig = 1.f / (1.f + __expf(-s));
    xc[id] = s * sig;
}

// ---------------------------------------------------------------------------
// dt = softplus(dt_raw + bias), in place
// ---------------------------------------------------------------------------
__global__ __launch_bounds__(256) void k_softplus_bias(float* __restrict__ dt,
                                                       const float* __restrict__ b) {
    int id = blockIdx.x * 256 + threadIdx.x;
    if (id >= SEQ_L * D_INNER) return;
    int d = id % D_INNER;
    float x = dt[id] + b[d];
    dt[id] = (x > 20.f) ? x : log1pf(__expf(x));
}

// ---------------------------------------------------------------------------
// Selective scan: one thread per channel d, 16 states in registers,
// sequential over L.  Fuses +xc*D and the SiLU(z) gate.
// ---------------------------------------------------------------------------
__global__ __launch_bounds__(256) void k_scan(const float* __restrict__ dt,
                                              const float* __restrict__ xc,
                                              const float* __restrict__ dbl,
                                              const float* __restrict__ A_log,
                                              const float* __restrict__ Dp,
                                              const float* __restrict__ xz,
                                              float* __restrict__ y) {
    int d = blockIdx.x * 256 + threadIdx.x;   // 6 blocks * 256 = 1536
    float a[D_STATE], s[D_STATE];
#pragma unroll
    for (int n = 0; n < D_STATE; ++n) {
        a[n] = -__expf(A_log[(size_t)d * D_STATE + n]);
        s[n] = 0.f;
    }
    float Dd = Dp[d];
    for (int l = 0; l < SEQ_L; ++l) {
        float dtv = dt[(size_t)l * D_INNER + d];
        float xv  = xc[(size_t)l * D_INNER + d];
        const float* bl = dbl + (size_t)l * 80;  // [0:48]=dt_in, [48:64]=B, [64:80]=C
        float yv = 0.f;
#pragma unroll
        for (int n = 0; n < D_STATE; ++n) {
            float dA = __expf(dtv * a[n]);
            s[n] = dA * s[n] + dtv * bl[DT_RANK + n] * xv;
            yv += s[n] * bl[DT_RANK + D_STATE + n];
        }
        float zv = xz[(size_t)l * (2 * D_INNER) + D_INNER + d];
        float sig = 1.f / (1.f + __expf(-zv));
        y[(size_t)l * D_INNER + d] = (yv + xv * Dd) * (zv * sig);
    }
}

// ---------------------------------------------------------------------------
// WMMA GEMM: C[M,N] (+= Acc) = A[M,K] * W[N,K]^T, f32 in/out, f16 compute.
// Each wave computes a 32x64 block (MT x NT 16x16 tiles), K-chunks of 32 via
// v_wmma_f32_16x16x32_f16.  A fragments reused across NT, B across MT.
// Fragment layouts used here (A/B in contiguous K order, not the ISA table order):
//   A: lane<16 -> M=lane, K = c+0..15;  lane>=16 -> M=lane-16, K = c+16..31
//   B: lane<16 -> row N=lane, K = c+0..15;          lane>=16 -> K = c+16..31
//   D: vgpr v: lanes0-15 M=v, lanes16-31 M=v+8, N=lane%16
// ---------------------------------------------------------------------------
static __device__ inline void packA16(v16h& f, const float* p) {
    float4 a0 = ((const float4*)p)[0];
    float4 a1 = ((const float4*)p)[1];
    float4 b0 = ((const float4*)(p + 8))[0];
    float4 b1 = ((const float4*)(p + 8))[1];
    f[0]=(_Float16)a0.x; f[1]=(_Float16)a0.y; f[2]=(_Float16)a0.z; f[3]=(_Float16)a0.w;
    f[4]=(_Float16)a1.x; f[5]=(_Float16)a1.y; f[6]=(_Float16)a1.z; f[7]=(_Float16)a1.w;
    f[8]=(_Float16)b0.x; f[9]=(_Float16)b0.y; f[10]=(_Float16)b0.z; f[11]=(_Float16)b0.w;
    f[12]=(_Float16)b1.x; f[13]=(_Float16)b1.y; f[14]=(_Float16)b1.z; f[15]=(_Float16)b1.w;
}
static __device__ inline void packB16(v16h& f, const float* p) {
    float4 a0 = ((const float4*)p)[0];
    float4 a1 = ((const float4*)p)[1];
    float4 a2 = ((const float4*)p)[2];
    float4 a3 = ((const float4*)p)[3];
    f[0]=(_Float16)a0.x; f[1]=(_Float16)a0.y; f[2]=(_Float16)a0.z; f[3]=(_Float16)a0.w;
    f[4]=(_Float16)a1.x; f[5]=(_Float16)a1.y; f[6]=(_Float16)a1.z; f[7]=(_Float16)a1.w;
    f[8]=(_Float16)a2.x; f[9]=(_Float16)a2.y; f[10]=(_Float16)a2.z; f[11]=(_Float16)a2.w;
    f[12]=(_Float16)a3.x; f[13]=(_Float16)a3.y; f[14]=(_Float16)a3.z; f[15]=(_Float16)a3.w;
}

__global__ __launch_bounds__(128) void k_gemm_wmma(
    const float* __restrict__ A, int lda,
    const float* __restrict__ B, int ldb,   // B is N x K row-major (weights)
    float* __restrict__ C, int ldc,
    const float* __restrict__ Acc,          // nullptr -> none (may alias C)
    int M, int N, int K) {
    const int lane = threadIdx.x & 31;
    const int wave = threadIdx.x >> 5;
    const int ntNB = (N + 16 * NT - 1) / (16 * NT);
    const int ntMB = M / (16 * MT);
    int w = blockIdx.x * 4 + wave;
    if (w >= ntMB * ntNB) return;
    const int bM = w / ntNB, bN = w % ntNB;
    const int row0 = bM * 16 * MT, col0 = bN * 16 * NT;

    const int  lh = lane & 15;
    const bool hi = lane >= 16;
    const int  aoff = hi ? 16 : 0;
    const int  boff = hi ? 16 : 0;

    const float* Am[MT];
#pragma unroll
    for (int i = 0; i < MT; ++i)
        Am[i] = A + (size_t)(row0 + 16 * i + lh) * lda;

    const float* Bn[NT];
    bool bv[NT];
#pragma unroll
    for (int j = 0; j < NT; ++j) {
        int bn = col0 + 16 * j + lh;
        bv[j] = (bn < N);
        Bn[j] = B + (size_t)(bv[j] ? bn : 0) * ldb;
    }

    v8f acc[MT][NT];
#pragma unroll
    for (int i = 0; i < MT; ++i)
#pragma unroll
        for (int j = 0; j < NT; ++j) acc[i][j] = (v8f){};

    const int  Kfull    = K & ~31;
    const bool interior = (col0 + 16 * NT <= N);   // uniform across wave

    if (interior) {
        // hot path: no guards anywhere in the loop
        for (int k0 = 0; k0 < Kfull; k0 += 32) {
            v16h af[MT], bf[NT];
#pragma unroll
            for (int i = 0; i < MT; ++i) packA16(af[i], Am[i] + k0 + aoff);
#pragma unroll
            for (int j = 0; j < NT; ++j) packB16(bf[j], Bn[j] + k0 + boff);
#pragma unroll
            for (int i = 0; i < MT; ++i)
#pragma unroll
                for (int j = 0; j < NT; ++j)
                    acc[i][j] = __builtin_amdgcn_wmma_f32_16x16x32_f16(
                        false, af[i], false, bf[j], (short)0, acc[i][j], false, false);
        }
    } else {
        // cold path: N-edge block, per-lane select on B columns
        for (int k0 = 0; k0 < Kfull; k0 += 32) {
            v16h af[MT], bf[NT];
#pragma unroll
            for (int i = 0; i < MT; ++i) packA16(af[i], Am[i] + k0 + aoff);
#pragma unroll
            for (int j = 0; j < NT; ++j) {
#pragma unroll
                for (int t = 0; t < 16; ++t)
                    bf[j][t] = (_Float16)(bv[j] ? Bn[j][k0 + boff + t] : 0.f);
            }
#pragma unroll
            for (int i = 0; i < MT; ++i)
#pragma unroll
                for (int j = 0; j < NT; ++j)
                    acc[i][j] = __builtin_amdgcn_wmma_f32_16x16x32_f16(
                        false, af[i], false, bf[j], (short)0, acc[i][j], false, false);
        }
    }

    if (Kfull < K) {   // K tail (K%32 == 16 case), runs once, fully guarded
        const int k0 = Kfull;
        v16h af[MT], bf[NT];
#pragma unroll
        for (int i = 0; i < MT; ++i) {
#pragma unroll
            for (int t = 0; t < 8; ++t) {
                int k1 = k0 + aoff + t, k2 = k0 + aoff + 8 + t;
                af[i][t]     = (_Float16)((k1 < K) ? Am[i][k1] : 0.f);
                af[i][8 + t] = (_Float16)((k2 < K) ? Am[i][k2] : 0.f);
            }
        }
#pragma unroll
        for (int j = 0; j < NT; ++j) {
#pragma unroll
            for (int t = 0; t < 16; ++t) {
                int k = k0 + boff + t;
                bf[j][t] = (_Float16)((bv[j] && k < K) ? Bn[j][k] : 0.f);
            }
        }
#pragma unroll
        for (int i = 0; i < MT; ++i)
#pragma unroll
            for (int j = 0; j < NT; ++j)
                acc[i][j] = __builtin_amdgcn_wmma_f32_16x16x32_f16(
                    false, af[i], false, bf[j], (short)0, acc[i][j], false, false);
    }

    // epilogue: D vgpr v -> row (v + hi*8), col = lane%16
    const int mb = hi ? 8 : 0;
#pragma unroll
    for (int i = 0; i < MT; ++i) {
#pragma unroll
        for (int j = 0; j < NT; ++j) {
            int n = col0 + 16 * j + lh;
            if (n < N) {
#pragma unroll
                for (int v = 0; v < 8; ++v) {
                    size_t idx = (size_t)(row0 + 16 * i + mb + v) * ldc + n;
                    float r = acc[i][j][v];
                    if (Acc) r += Acc[idx];
                    C[idx] = r;
                }
            }
        }
    }
}

static inline void gemm(const float* A, int lda, const float* B, int ldb,
                        float* C, int ldc, const float* Acc,
                        int M, int N, int K, hipStream_t s) {
    int waves = (M / (16 * MT)) * ((N + 16 * NT - 1) / (16 * NT));
    int blocks = (waves + 3) / 4;
    k_gemm_wmma<<<blocks, 128, 0, s>>>(A, lda, B, ldb, C, ldc, Acc, M, N, K);
}

// ---------------------------------------------------------------------------
extern "C" void kernel_launch(void* const* d_in, const int* in_sizes, int n_in,
                              void* d_out, int out_size, void* d_ws, size_t ws_size,
                              hipStream_t stream) {
    const int*   ids   = (const int*)  d_in[0];
    const float* emb   = (const float*)d_in[1];
    const float* in_w  = (const float*)d_in[2];
    const float* cw    = (const float*)d_in[3];
    const float* cb    = (const float*)d_in[4];
    const float* xw    = (const float*)d_in[5];
    const float* dtw   = (const float*)d_in[6];
    const float* dtb   = (const float*)d_in[7];
    const float* A_log = (const float*)d_in[8];
    const float* Dp    = (const float*)d_in[9];
    const float* ow    = (const float*)d_in[10];
    const float* nw    = (const float*)d_in[11];
    const float* nfw   = (const float*)d_in[12];

    float* ws  = (float*)d_ws;
    float* h   = ws;                        // 1024*768
    float* hn  = h   + SEQ_L * D_MODEL;     // 1024*768
    float* xz  = hn  + SEQ_L * D_MODEL;     // 1024*3072
    float* xc  = xz  + SEQ_L * 2 * D_INNER; // 1024*1536
    float* dbl = xc  + SEQ_L * D_INNER;     // 1024*80
    float* dtv = dbl + SEQ_L * 80;          // 1024*1536
    float* y   = dtv + SEQ_L * D_INNER;     // 1024*1536
    float* out = (float*)d_out;             // 1024*50280

    k_embed<<<SEQ_L, 256, 0, stream>>>(ids, emb, h);

    for (int i = 0; i < 4; ++i) {
        k_rmsnorm<<<SEQ_L, 256, 0, stream>>>(h, nw + (size_t)i * D_MODEL, hn, D_MODEL);
        // xz = hn @ in_proj_w^T   (1024 x 3072, K=768)
        gemm(hn, D_MODEL, in_w + (size_t)i * 2 * D_INNER * D_MODEL, D_MODEL,
             xz, 2 * D_INNER, nullptr, SEQ_L, 2 * D_INNER, D_MODEL, stream);
        // causal depthwise conv + silu
        k_conv_silu<<<(SEQ_L * D_INNER) / 256, 256, 0, stream>>>(
            xz, cw + (size_t)i * D_INNER * 4, cb + (size_t)i * D_INNER, xc);
        // dbl = xc @ x_proj_w^T   (1024 x 80, K=1536)
        gemm(xc, D_INNER, xw + (size_t)i * (DT_RANK + 2 * D_STATE) * D_INNER, D_INNER,
             dbl, DT_RANK + 2 * D_STATE, nullptr,
             SEQ_L, DT_RANK + 2 * D_STATE, D_INNER, stream);
        // dt_raw = dbl[:, :48] @ dt_proj_w^T   (1024 x 1536, K=48)
        gemm(dbl, DT_RANK + 2 * D_STATE, dtw + (size_t)i * D_INNER * DT_RANK, DT_RANK,
             dtv, D_INNER, nullptr, SEQ_L, D_INNER, DT_RANK, stream);
        k_softplus_bias<<<(SEQ_L * D_INNER) / 256, 256, 0, stream>>>(
            dtv, dtb + (size_t)i * D_INNER);
        // sequential selective scan + gate
        k_scan<<<D_INNER / 256, 256, 0, stream>>>(
            dtv, xc, dbl, A_log + (size_t)i * D_INNER * D_STATE,
            Dp + (size_t)i * D_INNER, xz, y);
        // h = h + y @ out_proj_w^T   (1024 x 768, K=1536), residual fused
        gemm(y, D_INNER, ow + (size_t)i * D_MODEL * D_INNER, D_INNER,
             h, D_MODEL, h, SEQ_L, D_MODEL, D_INNER, stream);
    }

    k_rmsnorm<<<SEQ_L, 256, 0, stream>>>(h, nfw, hn, D_MODEL);
    // logits = hn @ embedding^T   (1024 x 50280, K=768)
    gemm(hn, D_MODEL, emb, D_MODEL, out, 50280, nullptr, SEQ_L, 50280, D_MODEL, stream);
}
